// Self_ATT_RTree_41961830482186
// MI455X (gfx1250) — matrix operand order repaired and checked by reading, not compiled, not run on hardware
//
#include <hip/hip_runtime.h>
#include <hip/hip_bf16.h>
#include <math.h>

#define B_  16
#define S_  2048
#define N_  2048
#define P_  1024
#define H_  256
#define H2_ 512

typedef __attribute__((ext_vector_type(16))) __bf16 v16bf;
typedef __attribute__((ext_vector_type(8)))  __bf16 v8bf;
typedef __attribute__((ext_vector_type(8)))  float  v8f;

__device__ __forceinline__ v8f wmma_bf16(v16bf a, v16bf b, v8f c) {
  // D = A(16x32 bf16) x B(32x16 bf16) + C(16x16 f32)
  return __builtin_amdgcn_wmma_f32_16x16x32_bf16(
      /*neg_a=*/false, a, /*neg_b=*/false, b,
      /*c_mod=*/(short)0, c, /*reuse_a=*/false, /*reuse_b=*/false);
}

__device__ __forceinline__ v8f vzero8() {
  v8f z;
#pragma unroll
  for (int t = 0; t < 8; ++t) z[t] = 0.0f;
  return z;
}

__device__ __forceinline__ v16bf cat8(v8bf c0, v8bf c1) {
  v16bf r;
#pragma unroll
  for (int t = 0; t < 8; ++t) { r[t] = c0[t]; r[8 + t] = c1[t]; }
  return r;
}

// Pre-split bf16 fragment loads (16B b128 loads).
// A-operand (16x32, MxK): lane = row, chunks at k+8*half and k+16+8*half.
__device__ __forceinline__ v16bf ldA(const __bf16* __restrict__ row, int k0, int half) {
  return cat8(*(const v8bf*)(row + k0 + 8 * half),
              *(const v8bf*)(row + k0 + 16 + 8 * half));
}
// B-operand (32x16, KxN): lane = column, chunks at k+8*half and k+16+8*half.
__device__ __forceinline__ v16bf ldB(const __bf16* __restrict__ row, int k0, int half) {
  return cat8(*(const v8bf*)(row + k0 + 8 * half),
              *(const v8bf*)(row + k0 + 16 + 8 * half));
}

// f32 -> bf16 hi/lo split of 8 consecutive floats (used only where A is f32: attn).
__device__ __forceinline__ void cvt8(const float* __restrict__ p,
                                     v16bf& hi, v16bf& lo, int base) {
  float4 u = *(const float4*)(p);
  float4 v = *(const float4*)(p + 4);
  float x[8] = {u.x, u.y, u.z, u.w, v.x, v.y, v.z, v.w};
#pragma unroll
  for (int t = 0; t < 8; ++t) {
    __bf16 h = (__bf16)x[t];
    hi[base + t] = h;
    lo[base + t] = (__bf16)(x[t] - (float)h);
  }
}
__device__ __forceinline__ void loadA_f32(const float* __restrict__ row, int k0, int half,
                                          v16bf& hi, v16bf& lo) {
  cvt8(row + k0 + 8 * half, hi, lo, 0);
  cvt8(row + k0 + 16 + 8 * half, hi, lo, 8);
}

// One K-step, 32(M)x64(N) tile, all operands pre-split. 24 WMMAs, B transient.
__device__ __forceinline__ void gemm_step_pre(
    const __bf16* __restrict__ a0h, const __bf16* __restrict__ a0l,
    const __bf16* __restrict__ a1h, const __bf16* __restrict__ a1l,
    const __bf16* const* __restrict__ bh, const __bf16* const* __restrict__ bl,
    int k0, int half, v8f acc[2][4]) {
  v16bf a0hi = ldA(a0h, k0, half), a0lo = ldA(a0l, k0, half);
  v16bf a1hi = ldA(a1h, k0, half), a1lo = ldA(a1l, k0, half);
#pragma unroll
  for (int j = 0; j < 4; ++j) {
    v16bf bhi = ldB(bh[j], k0, half);
    v16bf blo = ldB(bl[j], k0, half);
    acc[0][j] = wmma_bf16(a0hi, bhi, acc[0][j]);
    acc[1][j] = wmma_bf16(a1hi, bhi, acc[1][j]);
    acc[0][j] = wmma_bf16(a0hi, blo, acc[0][j]);
    acc[1][j] = wmma_bf16(a1hi, blo, acc[1][j]);
    acc[0][j] = wmma_bf16(a0lo, bhi, acc[0][j]);
    acc[1][j] = wmma_bf16(a1lo, bhi, acc[1][j]);
  }
}

// Same but A is f32 (attn rows), converted in-registers.
__device__ __forceinline__ void gemm_step_mix(
    const float* __restrict__ a0, const float* __restrict__ a1,
    const __bf16* const* __restrict__ bh, const __bf16* const* __restrict__ bl,
    int k0, int half, v8f acc[2][4]) {
  v16bf a0hi, a0lo, a1hi, a1lo;
  loadA_f32(a0, k0, half, a0hi, a0lo);
  loadA_f32(a1, k0, half, a1hi, a1lo);
#pragma unroll
  for (int j = 0; j < 4; ++j) {
    v16bf bhi = ldB(bh[j], k0, half);
    v16bf blo = ldB(bl[j], k0, half);
    acc[0][j] = wmma_bf16(a0hi, bhi, acc[0][j]);
    acc[1][j] = wmma_bf16(a1hi, bhi, acc[1][j]);
    acc[0][j] = wmma_bf16(a0hi, blo, acc[0][j]);
    acc[1][j] = wmma_bf16(a1hi, blo, acc[1][j]);
    acc[0][j] = wmma_bf16(a0lo, bhi, acc[0][j]);
    acc[1][j] = wmma_bf16(a1lo, bhi, acc[1][j]);
  }
}

// ---------------------------------------------------------------- prep kernels
// elementwise f32 -> bf16 hi/lo split
__global__ void split_kernel(const float* __restrict__ src, __bf16* __restrict__ hi,
                             __bf16* __restrict__ lo, int n) {
  int i = blockIdx.x * 256 + threadIdx.x;
  if (i >= n) return;
  float x = src[i];
  __bf16 h = (__bf16)x;
  hi[i] = h;
  lo[i] = (__bf16)(x - (float)h);
}

// gather Q = output[b, num_pos[b,p], :] and split (B,P,H)
__global__ void gather_split_kernel(const float* __restrict__ outp,
                                    const int* __restrict__ num_pos,
                                    __bf16* __restrict__ hi, __bf16* __restrict__ lo) {
  int i = blockIdx.x * 256 + threadIdx.x;   // over B*P*H
  int r = i >> 8;                           // / H_
  int h = i & (H_ - 1);
  int b = r >> 10;                          // / P_
  int np = num_pos[r];
  float x = outp[((size_t)b * S_ + (size_t)np) * H_ + h];
  __bf16 hh = (__bf16)x;
  hi[i] = hh;
  lo[i] = (__bf16)(x - (float)hh);
}

// CT[b][h][n] = split(context[b][n][h])  (LDS-tiled 32x32)
__global__ void transpose_split_kernel(const float* __restrict__ ctx,
                                       __bf16* __restrict__ cthi,
                                       __bf16* __restrict__ ctlo) {
  __shared__ float tile[32][33];
  const int b  = blockIdx.z;
  const int n0 = blockIdx.x * 32;
  const int h0 = blockIdx.y * 32;
  const float* src = ctx + (size_t)b * N_ * H_;
  const size_t dbase = (size_t)b * H_ * N_;
  const int tx = threadIdx.x, ty = threadIdx.y;   // 32 x 8
#pragma unroll
  for (int r = 0; r < 32; r += 8)
    tile[ty + r][tx] = src[(size_t)(n0 + ty + r) * H_ + (h0 + tx)];
  __syncthreads();
#pragma unroll
  for (int r = 0; r < 32; r += 8) {
    float x = tile[tx][ty + r];
    __bf16 hh = (__bf16)x;
    size_t di = dbase + (size_t)(h0 + ty + r) * N_ + (n0 + tx);
    cthi[di] = hh;
    ctlo[di] = (__bf16)(x - (float)hh);
  }
}

// ---------------------------------------------------------------- scores
// attn[b,p,n] = sum_h Q[b,p,h]*C[b,n,h]  (masked with -inf for n >= len[b])
__global__ void __launch_bounds__(32, 1)
scores_kernel(const __bf16* __restrict__ qhi, const __bf16* __restrict__ qlo,
              const __bf16* __restrict__ chi, const __bf16* __restrict__ clo,
              const int* __restrict__ lens, float* __restrict__ attn) {
  const int lane = threadIdx.x & 31;
  const int col  = lane & 15;
  const int half = lane >> 4;
  const int b  = blockIdx.z;
  const int p0 = blockIdx.x * 32;
  const int n0 = blockIdx.y * 64;
  const int len = lens[b];

  const __bf16 *a0h, *a0l, *a1h, *a1l;
  {
    size_t r0 = ((size_t)b * P_ + (size_t)(p0 + col)) * H_;
    a0h = qhi + r0;           a0l = qlo + r0;
    a1h = qhi + r0 + 16 * H_; a1l = qlo + r0 + 16 * H_;
  }
  const __bf16* bh[4];
  const __bf16* bl[4];
#pragma unroll
  for (int j = 0; j < 4; ++j) {
    size_t rb = ((size_t)b * N_ + (size_t)(n0 + j * 16 + col)) * H_;
    bh[j] = chi + rb;
    bl[j] = clo + rb;
  }

  v8f acc[2][4];
#pragma unroll
  for (int i = 0; i < 2; ++i)
#pragma unroll
    for (int j = 0; j < 4; ++j) acc[i][j] = vzero8();

#pragma unroll 2
  for (int k0 = 0; k0 < H_; k0 += 32)
    gemm_step_pre(a0h, a0l, a1h, a1l, bh, bl, k0, half, acc);

#pragma unroll
  for (int i = 0; i < 2; ++i)
#pragma unroll
    for (int j = 0; j < 4; ++j) {
      int n = n0 + j * 16 + col;
      bool valid = n < len;
      float* o = attn + ((size_t)b * P_ + (size_t)(p0 + i * 16 + 8 * half)) * N_ + n;
#pragma unroll
      for (int r = 0; r < 8; ++r)
        o[(size_t)r * N_] = valid ? acc[i][j][r] : -__builtin_inff();
    }
}

// ---------------------------------------------------------------- softmax
__global__ void softmax_kernel(float* __restrict__ attn) {
  const int row = blockIdx.x;            // 0 .. B*P-1
  float* a = attn + (size_t)row * N_;
  __shared__ float red[256];
  const int tid = threadIdx.x;

  float mx = -__builtin_inff();
  float v[8];
#pragma unroll
  for (int t = 0; t < 8; ++t) {
    v[t] = a[tid + t * 256];
    mx = fmaxf(mx, v[t]);
  }
  red[tid] = mx;
  __syncthreads();
  for (int s = 128; s > 0; s >>= 1) {
    if (tid < s) red[tid] = fmaxf(red[tid], red[tid + s]);
    __syncthreads();
  }
  mx = red[0];
  __syncthreads();

  float sum = 0.0f;
#pragma unroll
  for (int t = 0; t < 8; ++t) {
    v[t] = expf(v[t] - mx);              // exp(-inf) -> 0, matches masked softmax
    sum += v[t];
  }
  red[tid] = sum;
  __syncthreads();
  for (int s = 128; s > 0; s >>= 1) {
    if (tid < s) red[tid] += red[tid + s];
    __syncthreads();
  }
  const float inv = 1.0f / red[0];
#pragma unroll
  for (int t = 0; t < 8; ++t) a[tid + t * 256] = v[t] * inv;
}

// ---------------------------------------------------------------- mix
// mix[b,p,h] = sum_n attn[b,p,n]*C[b,n,h]; writes pre-split bf16 hi/lo
__global__ void __launch_bounds__(32, 1)
mix_kernel(const float* __restrict__ attn,
           const __bf16* __restrict__ cthi, const __bf16* __restrict__ ctlo,
           __bf16* __restrict__ mixhi, __bf16* __restrict__ mixlo) {
  const int lane = threadIdx.x & 31;
  const int col  = lane & 15;
  const int half = lane >> 4;
  const int b  = blockIdx.z;
  const int p0 = blockIdx.x * 32;
  const int h0 = blockIdx.y * 64;

  const float* a0 = attn + ((size_t)b * P_ + (size_t)(p0 + col)) * N_;
  const float* a1 = a0 + (size_t)16 * N_;
  const __bf16* bh[4];
  const __bf16* bl[4];
#pragma unroll
  for (int j = 0; j < 4; ++j) {
    size_t rb = ((size_t)b * H_ + (size_t)(h0 + j * 16 + col)) * N_;
    bh[j] = cthi + rb;
    bl[j] = ctlo + rb;
  }

  v8f acc[2][4];
#pragma unroll
  for (int i = 0; i < 2; ++i)
#pragma unroll
    for (int j = 0; j < 4; ++j) acc[i][j] = vzero8();

#pragma unroll 2
  for (int k0 = 0; k0 < N_; k0 += 32)
    gemm_step_mix(a0, a1, bh, bl, k0, half, acc);

#pragma unroll
  for (int i = 0; i < 2; ++i)
#pragma unroll
    for (int j = 0; j < 4; ++j) {
      int h = h0 + j * 16 + col;
      size_t base = ((size_t)b * P_ + (size_t)(p0 + i * 16 + 8 * half)) * H_ + h;
#pragma unroll
      for (int r = 0; r < 8; ++r) {
        float x = acc[i][j][r];
        __bf16 hh = (__bf16)x;
        mixhi[base + (size_t)r * H_] = hh;
        mixlo[base + (size_t)r * H_] = (__bf16)(x - (float)hh);
      }
    }
}

// ---------------------------------------------------------------- final
// out[r,h] = tanh( sum_k [mix|Q][r,k] * W[h,k] + bias[h] ), r = b*P+p
__global__ void __launch_bounds__(32, 1)
final_kernel(const __bf16* __restrict__ mixhi, const __bf16* __restrict__ mixlo,
             const __bf16* __restrict__ qhi, const __bf16* __restrict__ qlo,
             const __bf16* __restrict__ whi, const __bf16* __restrict__ wlo,
             const float* __restrict__ bias, float* __restrict__ dst) {
  const int lane = threadIdx.x & 31;
  const int col  = lane & 15;
  const int half = lane >> 4;
  const int r0 = blockIdx.x * 32;
  const int h0 = blockIdx.y * 64;

  const size_t ra = (size_t)(r0 + col) * H_;
  const __bf16 *m0h = mixhi + ra, *m0l = mixlo + ra;
  const __bf16 *m1h = m0h + 16 * H_, *m1l = m0l + 16 * H_;
  const __bf16 *g0h = qhi + ra, *g0l = qlo + ra;
  const __bf16 *g1h = g0h + 16 * H_, *g1l = g0l + 16 * H_;

  const __bf16* bh[4];
  const __bf16* bl[4];
#pragma unroll
  for (int j = 0; j < 4; ++j) {
    size_t rb = (size_t)(h0 + j * 16 + col) * H2_;
    bh[j] = whi + rb;
    bl[j] = wlo + rb;
  }

  v8f acc[2][4];
#pragma unroll
  for (int i = 0; i < 2; ++i)
#pragma unroll
    for (int j = 0; j < 4; ++j) acc[i][j] = vzero8();

  // K = 0..255 from mix; W columns 0..255
#pragma unroll 2
  for (int k0 = 0; k0 < H_; k0 += 32)
    gemm_step_pre(m0h, m0l, m1h, m1l, bh, bl, k0, half, acc);
  // K = 256..511 from gathered Q; W columns 256..511
#pragma unroll
  for (int j = 0; j < 4; ++j) { bh[j] += H_; bl[j] += H_; }
#pragma unroll 2
  for (int k0 = 0; k0 < H_; k0 += 32)
    gemm_step_pre(g0h, g0l, g1h, g1l, bh, bl, k0, half, acc);

#pragma unroll
  for (int j = 0; j < 4; ++j) {
    int h = h0 + j * 16 + col;
    float bb = bias[h];
#pragma unroll
    for (int i = 0; i < 2; ++i) {
      float* o = dst + (size_t)(r0 + i * 16 + 8 * half) * H_ + h;
#pragma unroll
      for (int r = 0; r < 8; ++r)
        o[(size_t)r * H_] = tanhf(acc[i][j][r] + bb);
    }
  }
}

// ---------------------------------------------------------------- launch
extern "C" void kernel_launch(void* const* d_in, const int* in_sizes, int n_in,
                              void* d_out, int out_size, void* d_ws, size_t ws_size,
                              hipStream_t stream) {
  const float* outp  = (const float*)d_in[0];   // output  (B,S,H)
  const float* ctx   = (const float*)d_in[1];   // context (B,N,H)
  const float* W     = (const float*)d_in[2];   // W_out   (H,2H)
  const float* bias  = (const float*)d_in[3];   // b_out   (H,)
  const int*   npos  = (const int*)d_in[4];     // num_pos (B,P)
  const int*   lens  = (const int*)d_in[5];     // input_lengths (B,)

  float* out_dst = (float*)d_out;                       // (B,P,H)
  float* attn    = out_dst + (size_t)B_ * P_ * H_;      // (B,P,N)

  // workspace layout (bf16 planes), ~97 MB total
  char* w = (char*)d_ws;
  const size_t nC = (size_t)B_ * N_ * H_;   // 8.4M
  const size_t nQ = (size_t)B_ * P_ * H_;   // 4.2M
  const size_t nW = (size_t)H_ * H2_;       // 131K
  __bf16* Chi  = (__bf16*)w;            w += nC * 2;
  __bf16* Clo  = (__bf16*)w;            w += nC * 2;
  __bf16* CThi = (__bf16*)w;            w += nC * 2;
  __bf16* CTlo = (__bf16*)w;            w += nC * 2;
  __bf16* Qhi  = (__bf16*)w;            w += nQ * 2;
  __bf16* Qlo  = (__bf16*)w;            w += nQ * 2;
  __bf16* MXhi = (__bf16*)w;            w += nQ * 2;
  __bf16* MXlo = (__bf16*)w;            w += nQ * 2;
  __bf16* Whi  = (__bf16*)w;            w += nW * 2;
  __bf16* Wlo  = (__bf16*)w;            w += nW * 2;

  split_kernel<<<dim3((int)(nC / 256)), 256, 0, stream>>>(ctx, Chi, Clo, (int)nC);
  split_kernel<<<dim3((int)(nW / 256)), 256, 0, stream>>>(W, Whi, Wlo, (int)nW);
  gather_split_kernel<<<dim3((int)(nQ / 256)), 256, 0, stream>>>(outp, npos, Qhi, Qlo);
  transpose_split_kernel<<<dim3(N_ / 32, H_ / 32, B_), dim3(32, 8), 0, stream>>>(ctx, CThi, CTlo);

  scores_kernel<<<dim3(P_ / 32, N_ / 64, B_), 32, 0, stream>>>(Qhi, Qlo, Chi, Clo, lens, attn);
  softmax_kernel<<<dim3(B_ * P_), 256, 0, stream>>>(attn);
  mix_kernel<<<dim3(P_ / 32, H_ / 64, B_), 32, 0, stream>>>(attn, CThi, CTlo, MXhi, MXlo);
  final_kernel<<<dim3((B_ * P_) / 32, H_ / 64), 32, 0, stream>>>(MXhi, MXlo, Qhi, Qlo,
                                                                 Whi, Wlo, bias, out_dst);
}
